// MolDiffConsistencyModule_88862873355018
// MI455X (gfx1250) — matrix operand order repaired and checked by reading, not compiled, not run on hardware
//
#include <hip/hip_runtime.h>

typedef __attribute__((ext_vector_type(16))) _Float16 v16h;
typedef __attribute__((ext_vector_type(8)))  _Float16 v8h;
typedef __attribute__((ext_vector_type(8)))  float    v8f;

#define NNODE 4096
#define HID   256
#define NMOL  64
#define MATM  64
#define NEL   119
#define NBD   5

// ---------------- helpers ----------------
__device__ __forceinline__ float gelu_f(float x) {
    return 0.5f * x * (1.0f + erff(x * 0.70710678118654752f));
}
__device__ __forceinline__ float wave_sum(float v) {
#pragma unroll
    for (int m = 16; m >= 1; m >>= 1) v += __shfl_xor(v, m, 32);
    return v;
}
__device__ __forceinline__ float wave_max(float v) {
#pragma unroll
    for (int m = 16; m >= 1; m >>= 1) v = fmaxf(v, __shfl_xor(v, m, 32));
    return v;
}

// A fragment: 16x32 f16, lane L holds row m=L%16; halves = K {0..7,16..23} (L<16) / {8..15,24..31}
__device__ __forceinline__ v16h load_frag_a(const _Float16* base, int ld, int row_base,
                                            int k_base, int lane) {
    int m  = row_base + (lane & 15);
    int k0 = k_base + ((lane & 16) ? 8 : 0);
    const _Float16* p = base + (size_t)m * ld + k0;
    union { v16h v; v8h h[2]; } u;
    u.h[0] = *(const v8h*)(p);
    u.h[1] = *(const v8h*)(p + 16);
    return u.v;
}
// B fragment from transposed weights Wt[n][k]: lane L holds col n=L%16; halves = K {0..7,16..23} (L<16) / {8..15,24..31}
__device__ __forceinline__ v16h load_frag_b(const _Float16* bt, int ld, int n_base,
                                            int k_base, int lane) {
    int n  = n_base + (lane & 15);
    int k0 = k_base + ((lane & 16) ? 8 : 0);
    const _Float16* p = bt + (size_t)n * ld + k0;
    union { v16h v; v8h h[2]; } u;
    u.h[0] = *(const v8h*)(p);
    u.h[1] = *(const v8h*)(p + 16);
    return u.v;
}

// ---------------- conversion kernels ----------------
__global__ void k_f32_to_f16(const float* __restrict__ src, _Float16* __restrict__ dst, int n) {
    int i = blockIdx.x * blockDim.x + threadIdx.x;
    if (i < n) dst[i] = (_Float16)src[i];
}
// W [Ks][Ns] row-major (row stride Ns) -> Wt [Np][Kp] f16, zero-padded
__global__ void k_transpose_f16(const float* __restrict__ W, _Float16* __restrict__ Wt,
                                int Ks, int Ns, int Kp, int Np) {
    int idx = blockIdx.x * blockDim.x + threadIdx.x;
    if (idx >= Np * Kp) return;
    int n = idx / Kp, k = idx % Kp;
    float v = (k < Ks && n < Ns) ? W[(size_t)k * Ns + n] : 0.0f;
    Wt[idx] = (_Float16)v;
}

// ---------------- generic WMMA GEMM ----------------
// C[M,N] = act(A[M,K] @ W[K,N] + bias). A f16 row-major, W as Wt[n][k] f16.
// Block = 128 threads = 4 waves; wave w covers rows (blockIdx.x*4+w)*32 .. +31, cols blockIdx.y*64 .. +63.
__global__ __launch_bounds__(128)
void k_gemm_wmma(const _Float16* __restrict__ A, int lda,
                 const _Float16* __restrict__ Bt, int ldb,
                 const float* __restrict__ bias,
                 void* __restrict__ C, int ldc,
                 int K, int n_valid, int act, int out_half) {
    int lane = threadIdx.x & 31;
    int w    = threadIdx.x >> 5;
    int rowblk = (blockIdx.x * 4 + w) * 32;
    int colblk = blockIdx.y * 64;

    v8f acc[2][4] = {};
    int ksteps = K >> 5;
    for (int kt = 0; kt < ksteps; ++kt) {
        int kbase = kt * 32;
        v16h a0 = load_frag_a(A, lda, rowblk,      kbase, lane);
        v16h a1 = load_frag_a(A, lda, rowblk + 16, kbase, lane);
#pragma unroll
        for (int ct = 0; ct < 4; ++ct) {
            v16h b = load_frag_b(Bt, ldb, colblk + ct * 16, kbase, lane);
            acc[0][ct] = __builtin_amdgcn_wmma_f32_16x16x32_f16(false, a0, false, b,
                                                                (short)0, acc[0][ct], false, false);
            acc[1][ct] = __builtin_amdgcn_wmma_f32_16x16x32_f16(false, a1, false, b,
                                                                (short)0, acc[1][ct], false, false);
        }
    }
#pragma unroll
    for (int rt = 0; rt < 2; ++rt) {
#pragma unroll
        for (int ct = 0; ct < 4; ++ct) {
            int col = colblk + ct * 16 + (lane & 15);
            if (col >= n_valid) continue;
            float bv = bias ? bias[col] : 0.0f;
            int m0 = rowblk + rt * 16 + ((lane & 16) ? 8 : 0);
#pragma unroll
            for (int r = 0; r < 8; ++r) {
                float x = acc[rt][ct][r] + bv;
                if (act) x = gelu_f(x);
                if (out_half)
                    ((_Float16*)C)[(size_t)(m0 + r) * ldc + col] = (_Float16)x;
                else
                    ((float*)C)[(size_t)(m0 + r) * ldc + col] = x;
            }
        }
    }
}

// ---------------- layernorm: f32 in -> f16 out, wave per row, D=256 ----------------
__global__ void k_layernorm(const float* __restrict__ src, const float* __restrict__ g,
                            const float* __restrict__ b, _Float16* __restrict__ dst, int D) {
    int row  = blockIdx.x * (blockDim.x >> 5) + (threadIdx.x >> 5);
    int lane = threadIdx.x & 31;
    const float* x = src + (size_t)row * D;
    int per = D >> 5;
    float v[8];
    float s = 0.0f;
    for (int t = 0; t < per; ++t) { float y = x[t * 32 + lane]; v[t] = y; s += y; }
    s = wave_sum(s);
    float mu = s / (float)D;
    float s2 = 0.0f;
    for (int t = 0; t < per; ++t) { float d = v[t] - mu; s2 += d * d; }
    s2 = wave_sum(s2);
    float inv = rsqrtf(s2 / (float)D + 1e-5f);
    _Float16* o = dst + (size_t)row * D;
    for (int t = 0; t < per; ++t) {
        int c = t * 32 + lane;
        o[c] = (_Float16)((v[t] - mu) * inv * g[c] + b[c]);
    }
}

// ---------------- atom softmax + argmax + concat build ----------------
// logits [N,119] f32 -> writes vin[row][0:256]=hf16, [256:375]=probs f16, [375:384]=0;
// radii[row], maxallow[row] from argmax.
__global__ void k_atom_softmax(const float* __restrict__ logits, const _Float16* __restrict__ hf16,
                               _Float16* __restrict__ vin, float* __restrict__ radii,
                               float* __restrict__ maxallow, const float* __restrict__ cov,
                               const float* __restrict__ mv) {
    int row  = blockIdx.x * (blockDim.x >> 5) + (threadIdx.x >> 5);
    int lane = threadIdx.x & 31;
    const float* L = logits + (size_t)row * NEL;
    float lv[4];
    float mx = -1e30f; int mi = 0;
#pragma unroll
    for (int t = 0; t < 4; ++t) {
        int c = t * 32 + lane;
        float v = (c < NEL) ? L[c] : -1e30f;
        lv[t] = v;
        if (v > mx) { mx = v; mi = c; }   // increasing idx -> first-max kept
    }
    float rmx = wave_max(mx);
    // argmax reduce (earliest index on ties)
    float av = mx; int ai = mi;
#pragma unroll
    for (int m = 16; m >= 1; m >>= 1) {
        float ov = __shfl_xor(av, m, 32);
        int   oi = __shfl_xor(ai, m, 32);
        if (ov > av || (ov == av && oi < ai)) { av = ov; ai = oi; }
    }
    float s = 0.0f;
#pragma unroll
    for (int t = 0; t < 4; ++t) {
        int c = t * 32 + lane;
        if (c < NEL) { lv[t] = expf(lv[t] - rmx); s += lv[t]; }
    }
    s = wave_sum(s);
    float inv = 1.0f / s;
    _Float16* vrow = vin + (size_t)row * 384;
#pragma unroll
    for (int t = 0; t < 4; ++t) {
        int c = t * 32 + lane;
        if (c < 128) vrow[256 + c] = (_Float16)((c < NEL) ? lv[t] * inv : 0.0f);
    }
    const _Float16* hr = hf16 + (size_t)row * HID;
#pragma unroll
    for (int t = 0; t < 8; ++t) { int c = t * 32 + lane; vrow[c] = hr[c]; }
    if (lane == 0) {
        int p = ai;
        int pc = p < 118 ? p : 118;
        radii[row]    = cov[pc];
        maxallow[row] = mv[p];
    }
}

// ---------------- valency head: dot(t2[row], vW3) + b, softplus ----------------
__global__ void k_val(const _Float16* __restrict__ t2, const float* __restrict__ w3,
                      const float* __restrict__ b3, float* __restrict__ val) {
    int row  = blockIdx.x * (blockDim.x >> 5) + (threadIdx.x >> 5);
    int lane = threadIdx.x & 31;
    const _Float16* x = t2 + (size_t)row * 128;
    float s = 0.0f;
#pragma unroll
    for (int t = 0; t < 4; ++t) { int c = t * 32 + lane; s += (float)x[c] * w3[c]; }
    s = wave_sum(s);
    if (lane == 0) {
        float z = s + b3[0];
        val[row] = (z > 30.0f) ? z : log1pf(expf(z));
    }
}

// ---------------- valency_constrained: vc[i][j] = min(val[j], maxallow[i]) ----------------
__global__ void k_vfill(const float* __restrict__ val, const float* __restrict__ maxallow,
                        float* __restrict__ out) {
    int i  = blockIdx.y;
    int j4 = (blockIdx.x * blockDim.x + threadIdx.x) * 4;
    float ma = maxallow[i];
    float4 v = *(const float4*)(val + j4);
    float4 r = make_float4(fminf(v.x, ma), fminf(v.y, ma), fminf(v.z, ma), fminf(v.w, ma));
    *(float4*)(out + (size_t)i * NNODE + j4) = r;
}

// ---------------- bond predictor: one block per (molecule, i); 64 pair-rows ----------------
__global__ __launch_bounds__(256)
void k_bond(const float* __restrict__ P, const float* __restrict__ Q,
            const float* __restrict__ W4 /*[4][256]*/, const float* __restrict__ bb1,
            const float* __restrict__ bg, const float* __restrict__ bbn,
            const _Float16* __restrict__ bW2t, const float* __restrict__ bb2,
            const _Float16* __restrict__ bW3t, const float* __restrict__ bb3,
            const float* __restrict__ pos, const float* __restrict__ radii,
            float* __restrict__ outLogits, float* __restrict__ outAdj) {
    __shared__ _Float16 x1[64 * 264];   // 64 rows, padded stride
    __shared__ _Float16 x2[64 * 136];
    __shared__ float    lg[64 * 16];

    int bi   = blockIdx.x;
    int bmol = bi >> 6, i = bi & 63;
    int gi   = bmol * MATM + i;
    int tid  = threadIdx.x, lane = tid & 31, w = tid >> 5;

    float pix = pos[gi * 3 + 0], piy = pos[gi * 3 + 1], piz = pos[gi * 3 + 2];
    float ri  = radii[gi];

    // hoist per-column constants (i-row of P, scalar-feature weight rows, bias, LN params)
    const float* Pi = P + (size_t)gi * HID;
    float pv[8], w4a[8], w4b[8], w4c[8], w4d[8], b1v[8], gv[8], bnv[8];
#pragma unroll
    for (int t = 0; t < 8; ++t) {
        int c = t * 32 + lane;
        pv[t]  = Pi[c];
        w4a[t] = W4[c];        w4b[t] = W4[256 + c];
        w4c[t] = W4[512 + c];  w4d[t] = W4[768 + c];
        b1v[t] = bb1[c];       gv[t]  = bg[c];        bnv[t] = bbn[c];
    }

    // phase 1: build x1 rows (GELU then LN), wave w handles rows w*8..w*8+7
    for (int rr = 0; rr < 8; ++rr) {
        int j  = w * 8 + rr;
        int gj = bmol * MATM + j;
        float dx = pix - pos[gj * 3 + 0];
        float dy = piy - pos[gj * 3 + 1];
        float dz = piz - pos[gj * 3 + 2];
        float dist = sqrtf(dx * dx + dy * dy + dz * dz);
        float expd = ri + radii[gj];
        float ratio = dist / expd;
        float close = (dist < 3.0f) ? 1.0f : 0.0f;
        const float* Qj = Q + (size_t)gj * HID;
        float v[8];
        float s = 0.0f;
#pragma unroll
        for (int t = 0; t < 8; ++t) {
            int c = t * 32 + lane;
            float x = pv[t] + Qj[c] + dist * w4a[t] + expd * w4b[t]
                    + ratio * w4c[t] + close * w4d[t] + b1v[t];
            x = gelu_f(x);
            v[t] = x; s += x;
        }
        s = wave_sum(s);
        float mu = s * (1.0f / 256.0f);
        float s2 = 0.0f;
#pragma unroll
        for (int t = 0; t < 8; ++t) { float d = v[t] - mu; s2 += d * d; }
        s2 = wave_sum(s2);
        float inv = rsqrtf(s2 * (1.0f / 256.0f) + 1e-5f);
#pragma unroll
        for (int t = 0; t < 8; ++t) {
            int c = t * 32 + lane;
            x1[j * 264 + c] = (_Float16)((v[t] - mu) * inv * gv[t] + bnv[t]);
        }
    }
    __syncthreads();

    // phase 2: x1(64x256) @ bW2 -> GELU -> x2(64x128). 8 waves: rt=w&3, col half = w>>2
    {
        int rt = w & 3;
        int cb = (w >> 2) * 64;
        v8f acc[4] = {};
        for (int kt = 0; kt < 8; ++kt) {
            int kbase = kt * 32;
            v16h a = load_frag_a(x1, 264, rt * 16, kbase, lane);
#pragma unroll
            for (int ct = 0; ct < 4; ++ct) {
                v16h b = load_frag_b(bW2t, 256, cb + ct * 16, kbase, lane);
                acc[ct] = __builtin_amdgcn_wmma_f32_16x16x32_f16(false, a, false, b,
                                                                 (short)0, acc[ct], false, false);
            }
        }
#pragma unroll
        for (int ct = 0; ct < 4; ++ct) {
            int col = cb + ct * 16 + (lane & 15);
            int m0  = rt * 16 + ((lane & 16) ? 8 : 0);
            float bv = bb2[col];
#pragma unroll
            for (int r = 0; r < 8; ++r)
                x2[(m0 + r) * 136 + col] = (_Float16)gelu_f(acc[ct][r] + bv);
        }
    }
    __syncthreads();

    // phase 3: x2(64x128) @ bW3t(16x128) -> lg. waves 0..3, one 16-row tile each
    if (w < 4) {
        v8f acc = {};
        for (int kt = 0; kt < 4; ++kt) {
            int kbase = kt * 32;
            v16h a = load_frag_a(x2, 136, w * 16, kbase, lane);
            v16h b = load_frag_b(bW3t, 128, 0, kbase, lane);
            acc = __builtin_amdgcn_wmma_f32_16x16x32_f16(false, a, false, b,
                                                         (short)0, acc, false, false);
        }
        int m0 = w * 16 + ((lane & 16) ? 8 : 0);
        int n  = lane & 15;
#pragma unroll
        for (int r = 0; r < 8; ++r) lg[(m0 + r) * 16 + n] = acc[r];
    }
    __syncthreads();

    // phase 4: per-pair epilogue
    if (tid < 64) {
        int j = tid;
        float l[NBD], e[NBD];
        float mx = -1e30f;
#pragma unroll
        for (int t = 0; t < NBD; ++t) { l[t] = lg[j * 16 + t] + bb3[t]; mx = fmaxf(mx, l[t]); }
        float s = 0.0f;
#pragma unroll
        for (int t = 0; t < NBD; ++t) { e[t] = expf(l[t] - mx); s += e[t]; }
        float inv = 1.0f / s;
        float* adj = outAdj + ((size_t)(bmol * MATM + i) * MATM + j) * NBD;
        if (j == i) {
            adj[0] = 1.0f; adj[1] = 0.0f; adj[2] = 0.0f; adj[3] = 0.0f; adj[4] = 0.0f;
        } else {
#pragma unroll
            for (int t = 0; t < NBD; ++t) adj[t] = e[t] * inv;
            int jj = j - (j > i ? 1 : 0);
            float* bl = outLogits + ((size_t)bmol * (MATM * (MATM - 1)) + i * (MATM - 1) + jj) * NBD;
#pragma unroll
            for (int t = 0; t < NBD; ++t) bl[t] = l[t];
        }
    }
}

// ---------------- host launcher ----------------
extern "C" void kernel_launch(void* const* d_in, const int* in_sizes, int n_in,
                              void* d_out, int out_size, void* d_ws, size_t ws_size,
                              hipStream_t stream) {
    const float* h   = (const float*)d_in[0];
    const float* pos = (const float*)d_in[1];
    const float* aW1 = (const float*)d_in[4];  const float* ab1 = (const float*)d_in[5];
    const float* ag  = (const float*)d_in[6];  const float* abn = (const float*)d_in[7];
    const float* aW2 = (const float*)d_in[8];  const float* ab2 = (const float*)d_in[9];
    const float* aW3 = (const float*)d_in[10]; const float* ab3 = (const float*)d_in[11];
    const float* vW1 = (const float*)d_in[12]; const float* vb1 = (const float*)d_in[13];
    const float* vg  = (const float*)d_in[14]; const float* vbn = (const float*)d_in[15];
    const float* vW2 = (const float*)d_in[16]; const float* vb2 = (const float*)d_in[17];
    const float* vW3 = (const float*)d_in[18]; const float* vb3 = (const float*)d_in[19];
    const float* bW1 = (const float*)d_in[20]; const float* bb1 = (const float*)d_in[21];
    const float* bg  = (const float*)d_in[22]; const float* bbn = (const float*)d_in[23];
    const float* bW2 = (const float*)d_in[24]; const float* bb2 = (const float*)d_in[25];
    const float* bW3 = (const float*)d_in[26]; const float* bb3 = (const float*)d_in[27];
    const float* mv  = (const float*)d_in[28]; const float* cov = (const float*)d_in[29];

    // output layout (floats)
    float* out_logits = (float*)d_out;                               // [4096,119]
    float* out_bond   = out_logits + (size_t)NNODE * NEL;            // [258048,5]
    float* out_vc     = out_bond + (size_t)NMOL * MATM * (MATM - 1) * NBD; // [4096,4096]
    float* out_adj    = out_vc + (size_t)NNODE * NNODE;              // [64,64,64,5]

    // workspace layout
    char* ws = (char*)d_ws;
    size_t off = 0;
    auto alloc = [&](size_t bytes) -> void* {
        void* p = ws + off;
        off += (bytes + 255) & ~(size_t)255;
        return p;
    };
    _Float16* hf16  = (_Float16*)alloc((size_t)NNODE * HID * 2);
    _Float16* aW1t  = (_Float16*)alloc(256 * 256 * 2);
    _Float16* aW2t  = (_Float16*)alloc(128 * 256 * 2);
    _Float16* aW3t  = (_Float16*)alloc(128 * 128 * 2);
    _Float16* vW1t  = (_Float16*)alloc(256 * 384 * 2);
    _Float16* vW2t  = (_Float16*)alloc(128 * 256 * 2);
    _Float16* bW1at = (_Float16*)alloc(256 * 256 * 2);
    _Float16* bW1bt = (_Float16*)alloc(256 * 256 * 2);
    _Float16* bW2t  = (_Float16*)alloc(128 * 256 * 2);
    _Float16* bW3t  = (_Float16*)alloc(16 * 128 * 2);
    float*    gbuf  = (float*)alloc((size_t)NNODE * HID * 4);   // pre-LN f32 (reused)
    _Float16* lnbuf = (_Float16*)alloc((size_t)NNODE * HID * 2); // post-LN f16 (reused)
    _Float16* t2buf = (_Float16*)alloc((size_t)NNODE * 128 * 2); // layer2 out (reused)
    _Float16* vin   = (_Float16*)alloc((size_t)NNODE * 384 * 2);
    float*    val   = (float*)alloc(NNODE * 4);
    float*    radii = (float*)alloc(NNODE * 4);
    float*    mxal  = (float*)alloc(NNODE * 4);
    float*    Pbuf  = (float*)alloc((size_t)NNODE * HID * 4);
    float*    Qbuf  = (float*)alloc((size_t)NNODE * HID * 4);
    (void)ws_size; (void)in_sizes; (void)n_in; (void)out_size;

    // --- conversions ---
    k_f32_to_f16<<<(NNODE * HID + 255) / 256, 256, 0, stream>>>(h, hf16, NNODE * HID);
    k_transpose_f16<<<(256 * 256 + 255) / 256, 256, 0, stream>>>(aW1, aW1t, 256, 256, 256, 256);
    k_transpose_f16<<<(128 * 256 + 255) / 256, 256, 0, stream>>>(aW2, aW2t, 256, 128, 256, 128);
    k_transpose_f16<<<(128 * 128 + 255) / 256, 256, 0, stream>>>(aW3, aW3t, 128, 119, 128, 128);
    k_transpose_f16<<<(256 * 384 + 255) / 256, 256, 0, stream>>>(vW1, vW1t, 375, 256, 384, 256);
    k_transpose_f16<<<(128 * 256 + 255) / 256, 256, 0, stream>>>(vW2, vW2t, 256, 128, 256, 128);
    k_transpose_f16<<<(256 * 256 + 255) / 256, 256, 0, stream>>>(bW1, bW1at, 256, 256, 256, 256);
    k_transpose_f16<<<(256 * 256 + 255) / 256, 256, 0, stream>>>(bW1 + 256 * 256, bW1bt, 256, 256, 256, 256);
    k_transpose_f16<<<(128 * 256 + 255) / 256, 256, 0, stream>>>(bW2, bW2t, 256, 128, 256, 128);
    k_transpose_f16<<<(16 * 128 + 255) / 256, 256, 0, stream>>>(bW3, bW3t, 128, 5, 128, 16);

    // --- atom predictor ---
    k_gemm_wmma<<<dim3(NNODE / 128, 4), 128, 0, stream>>>(hf16, 256, aW1t, 256, ab1,
                                                          gbuf, 256, 256, 256, 1, 0);
    k_layernorm<<<NNODE / 8, 256, 0, stream>>>(gbuf, ag, abn, lnbuf, 256);
    k_gemm_wmma<<<dim3(NNODE / 128, 2), 128, 0, stream>>>(lnbuf, 256, aW2t, 256, ab2,
                                                          t2buf, 128, 256, 128, 1, 1);
    k_gemm_wmma<<<dim3(NNODE / 128, 2), 128, 0, stream>>>(t2buf, 128, aW3t, 128, ab3,
                                                          out_logits, NEL, 128, NEL, 0, 0);
    k_atom_softmax<<<NNODE / 8, 256, 0, stream>>>(out_logits, hf16, vin, radii, mxal, cov, mv);

    // --- valency predictor ---
    k_gemm_wmma<<<dim3(NNODE / 128, 4), 128, 0, stream>>>(vin, 384, vW1t, 384, vb1,
                                                          gbuf, 256, 384, 256, 1, 0);
    k_layernorm<<<NNODE / 8, 256, 0, stream>>>(gbuf, vg, vbn, lnbuf, 256);
    k_gemm_wmma<<<dim3(NNODE / 128, 2), 128, 0, stream>>>(lnbuf, 256, vW2t, 256, vb2,
                                                          t2buf, 128, 256, 128, 1, 1);
    k_val<<<NNODE / 8, 256, 0, stream>>>(t2buf, vW3, vb3, val);
    k_vfill<<<dim3(NNODE / 1024, NNODE), 256, 0, stream>>>(val, mxal, out_vc);

    // --- bond predictor ---
    k_gemm_wmma<<<dim3(NNODE / 128, 4), 128, 0, stream>>>(hf16, 256, bW1at, 256, (const float*)nullptr,
                                                          Pbuf, 256, 256, 256, 0, 0);
    k_gemm_wmma<<<dim3(NNODE / 128, 4), 128, 0, stream>>>(hf16, 256, bW1bt, 256, (const float*)nullptr,
                                                          Qbuf, 256, 256, 256, 0, 0);
    k_bond<<<NMOL * MATM, 256, 0, stream>>>(Pbuf, Qbuf, bW1 + 512 * 256, bb1, bg, bbn,
                                            bW2t, bb2, bW3t, bb3, pos, radii,
                                            out_bond, out_adj);
}
